// _NeuralSplineCouplingLayer_42812234006942
// MI455X (gfx1250) — matrix operand order repaired and checked by reading, not compiled, not run on hardware
//
#include <hip/hip_runtime.h>

// ---------------------------------------------------------------------------
// NeuralSplineCouplingLayer fused kernel for gfx1250 (MI455X, wave32, WMMA).
//   BATCH=65536, DIM=64, CTX=64, HID=512, K=16, N_T=32, N_P=47.
// v2: 32 rows per wave (two 16-row M-tiles share every B fragment) -> halves
// L2 weight traffic and doubles WMMA per vmem load; spline chunk = exactly
// one transform (48 padded cols), lane==row, no cross-lane logdet reduction.
// ---------------------------------------------------------------------------

typedef __attribute__((ext_vector_type(16))) _Float16 v16h;
typedef __attribute__((ext_vector_type(8)))  _Float16 v8h;
typedef __attribute__((ext_vector_type(8)))  float    v8f;

#define BATCH   65536
#define DIM     64
#define CTX     64
#define INP     128
#define HID     512
#define N_T     32
#define N_P     47
#define NPPAD   48
#define NOUTPAD (N_T * NPPAD)   // 1536
#define TB      5.0f

#define WAVES    4
#define ROWS_PW  32                          // 2 M-tiles per wave
#define ROWS_PB  (WAVES * ROWS_PW)           // 128
#define NBLOCKS  (BATCH / ROWS_PB)           // 512

// per-wave LDS layout (bytes)
#define LDS_INP  0                           // 32x128 f16 =  8192
#define LDS_H1   8192                        // 32x512 f16 = 32768
#define LDS_H2   40960                       // 32x512 f16 = 32768
#define LDS_RAW  73728                       // 32x48  f32 =  6144
#define WAVE_LDS 79872
#define DYN_LDS  (WAVE_LDS * WAVES)          // 319488 B  (<= 320KB WGP LDS)

// ---- WMMA fragment loads -------------------------------------------------
// A (16x32 f16): lane<16 -> row M=lane, K in {0..7}U{16..23};
//                lane>=16 -> row M=lane-16, K in {8..15}U{24..31}.
__device__ __forceinline__ v16h ldA(const _Float16* p) {
  v8h lo = *(const v8h*)(p);
  v8h hi = *(const v8h*)(p + 16);
  return __builtin_shufflevector(lo, hi, 0,1,2,3,4,5,6,7,8,9,10,11,12,13,14,15);
}
// B (32x16 f16): lane<16 -> col N=lane, K in {0..7}U{16..23}; lane>=16 -> K in {8..15}U{24..31}.
__device__ __forceinline__ v16h ldB(const _Float16* p) {
  v8h lo = *(const v8h*)(p);
  v8h hi = *(const v8h*)(p + 16);
  return __builtin_shufflevector(lo, hi, 0,1,2,3,4,5,6,7,8,9,10,11,12,13,14,15);
}

// 32xK (LDS, row stride `as`) x (K x 16) (global, transposed N-major, row
// stride K) -> two 16x16 f32 accumulators (rows 0-15 and 16-31).
__device__ __forceinline__ void gemm32(const _Float16* A, int as,
                                       const _Float16* Bt, int n, int K,
                                       int lane, v8f& acc0, v8f& acc1) {
  const int m   = lane & 15;
  const int k0a = (lane & 16) ? 8  : 0;
  const int k0b = (lane & 16) ? 8 : 0;
  const _Float16* arow0 = A + m * as + k0a;
  const _Float16* arow1 = arow0 + 16 * as;
  const _Float16* brow  = Bt + (size_t)n * K + k0b;
  // prefetch the same row of the NEXT 16-column tile (one panel ahead)
  __builtin_prefetch(Bt + (size_t)(n + 16) * K, 0, 1);
  for (int kb = 0; kb < K; kb += 32) {
    v16h b  = ldB(brow + kb);
    v16h a0 = ldA(arow0 + kb);
    v16h a1 = ldA(arow1 + kb);
    acc0 = __builtin_amdgcn_wmma_f32_16x16x32_f16(
        false, a0, false, b, (short)0, acc0, false, false);
    acc1 = __builtin_amdgcn_wmma_f32_16x16x32_f16(
        false, a1, false, b, (short)0, acc1, false, false);
  }
}

__device__ __forceinline__ float siluf(float v) {
  return v / (1.f + __expf(-v));
}
__device__ __forceinline__ float softplusf(float z) {
  return (z > 20.f) ? z : __logf(1.f + __expf(z));
}

// ---- weight conversion: f32 (K,N) -> f16 transposed (N,K); W3 padded -----
__global__ void nscl_convert_weights(const float* __restrict__ W1,
                                     const float* __restrict__ W2,
                                     const float* __restrict__ W3,
                                     _Float16* __restrict__ W1t,
                                     _Float16* __restrict__ W2t,
                                     _Float16* __restrict__ W3t) {
  int i = blockIdx.x * blockDim.x + threadIdx.x;
  int stride = gridDim.x * blockDim.x;
  for (int idx = i; idx < HID * INP; idx += stride) {
    int n = idx / INP, k = idx % INP;
    W1t[idx] = (_Float16)W1[k * HID + n];
  }
  for (int idx = i; idx < HID * HID; idx += stride) {
    int n = idx / HID, k = idx % HID;
    W2t[idx] = (_Float16)W2[k * HID + n];
  }
  for (int idx = i; idx < NOUTPAD * HID; idx += stride) {
    int np = idx / HID, k = idx % HID;
    int t = np / NPPAD, p = np % NPPAD;
    W3t[idx] = (p < N_P) ? (_Float16)W3[k * (N_T * N_P) + t * N_P + p]
                         : (_Float16)0.f;
  }
}

// ---- fused forward -------------------------------------------------------
__global__ void __launch_bounds__(WAVES * 32)
nscl_fused(const float* __restrict__ x, const float* __restrict__ ctx,
           const float* __restrict__ b1, const float* __restrict__ b2,
           const float* __restrict__ b3,
           const _Float16* __restrict__ W1t, const _Float16* __restrict__ W2t,
           const _Float16* __restrict__ W3t,
           float* __restrict__ y_out, float* __restrict__ logdet_out) {
  extern __shared__ char smem_raw[];
  const int lane = threadIdx.x & 31;
  const int wave = threadIdx.x >> 5;
  const int row0 = (blockIdx.x * WAVES + wave) * ROWS_PW;

  char* wbase = smem_raw + wave * WAVE_LDS;
  _Float16* inp  = (_Float16*)(wbase + LDS_INP);  // 32 x 128
  _Float16* h1   = (_Float16*)(wbase + LDS_H1);   // 32 x 512
  _Float16* h2   = (_Float16*)(wbase + LDS_H2);   // 32 x 512
  float*    rawT = (float*)(wbase + LDS_RAW);     // 32 x 48

  // ---- stage masked-x || context; write-through even y columns ----
  for (int idx = lane; idx < ROWS_PW * DIM; idx += 32) {
    int r = idx >> 6, c = idx & 63;
    float xv = x[(size_t)(row0 + r) * DIM + c];
    float cv = ctx[(size_t)(row0 + r) * CTX + c];
    inp[r * INP + c]       = (_Float16)((c & 1) ? 0.f : xv);
    inp[r * INP + DIM + c] = (_Float16)cv;
    if (!(c & 1)) y_out[(size_t)(row0 + r) * DIM + c] = xv;
  }
  __syncthreads();

  const int nlo = lane & 15;
  const int mof = (lane & 16) ? 8 : 0;

  // ---- GEMM1 + silu -> h1 ----
#pragma unroll 1
  for (int nt = 0; nt < HID / 16; ++nt) {
    int n = nt * 16 + nlo;
    v8f acc0 = {}, acc1 = {};
    gemm32(inp, INP, W1t, n, INP, lane, acc0, acc1);
    float bias = b1[n];
#pragma unroll
    for (int r = 0; r < 8; ++r) {
      h1[(r + mof) * HID + n]      = (_Float16)siluf(acc0[r] + bias);
      h1[(r + mof + 16) * HID + n] = (_Float16)siluf(acc1[r] + bias);
    }
  }
  __syncthreads();

  // ---- GEMM2 + silu -> h2 ----
#pragma unroll 1
  for (int nt = 0; nt < HID / 16; ++nt) {
    int n = nt * 16 + nlo;
    v8f acc0 = {}, acc1 = {};
    gemm32(h1, HID, W2t, n, HID, lane, acc0, acc1);
    float bias = b2[n];
#pragma unroll
    for (int r = 0; r < 8; ++r) {
      h2[(r + mof) * HID + n]      = (_Float16)siluf(acc0[r] + bias);
      h2[(r + mof + 16) * HID + n] = (_Float16)siluf(acc1[r] + bias);
    }
  }
  __syncthreads();

  // ---- GEMM3 chunked per transform + spline (lane == row) ----
  float logacc = 0.f;
#pragma unroll 1
  for (int t = 0; t < N_T; ++t) {
#pragma unroll 1
    for (int j = 0; j < 3; ++j) {        // 3 N-tiles = 48 padded params
      int p = j * 16 + nlo;              // padded param index in [0,48)
      int n = t * NPPAD + p;             // padded column in [0,1536)
      v8f acc0 = {}, acc1 = {};
      gemm32(h2, HID, W3t, n, HID, lane, acc0, acc1);
      float bias = (p < N_P) ? b3[t * N_P + p] : 0.f;
#pragma unroll
      for (int r = 0; r < 8; ++r) {
        rawT[(r + mof) * NPPAD + p]      = acc0[r] + bias;
        rawT[(r + mof + 16) * NPPAD + p] = acc1[r] + bias;
      }
    }
    __syncthreads();

    {
      const int row = lane;              // one spline problem per lane
      const float* P = rawT + row * NPPAD;
      float Pr[47];
#pragma unroll
      for (int i = 0; i < 47; ++i) Pr[i] = P[i];

      // softmax(P[0:16]) * 2B -> Wb
      float mx = Pr[0];
#pragma unroll
      for (int i = 1; i < 16; ++i) mx = fmaxf(mx, Pr[i]);
      float Wb[16], sw = 0.f;
#pragma unroll
      for (int i = 0; i < 16; ++i) { Wb[i] = __expf(Pr[i] - mx); sw += Wb[i]; }
      float scw = (2.f * TB) / sw;
#pragma unroll
      for (int i = 0; i < 16; ++i) Wb[i] *= scw;

      // softmax(P[16:32]) * 2B -> Hb
      mx = Pr[16];
#pragma unroll
      for (int i = 17; i < 32; ++i) mx = fmaxf(mx, Pr[i]);
      float Hb[16], sh = 0.f;
#pragma unroll
      for (int i = 0; i < 16; ++i) { Hb[i] = __expf(Pr[16 + i] - mx); sh += Hb[i]; }
      float sch = (2.f * TB) / sh;
#pragma unroll
      for (int i = 0; i < 16; ++i) Hb[i] *= sch;

      // derivatives: D = [1, softplus(P[32:47]), 1]
      float d[17];
      d[0] = 1.f; d[16] = 1.f;
#pragma unroll
      for (int i = 0; i < 15; ++i) d[i + 1] = softplusf(Pr[32 + i]);

      // cumulative knots
      float cw[17], ch[17];
      cw[0] = -TB; ch[0] = -TB;
#pragma unroll
      for (int i = 0; i < 16; ++i) {
        cw[i + 1] = cw[i] + Wb[i];
        ch[i + 1] = ch[i] + Hb[i];
      }

      float xt = x[(size_t)(row0 + row) * DIM + 2 * t + 1];
      float xc = fminf(fmaxf(xt, -TB + 1e-6f), TB - 1e-6f);

      int bin = 0;
#pragma unroll
      for (int i = 1; i <= 16; ++i) bin += (cw[i] < xc) ? 1 : 0;
      bin = (bin > 15) ? 15 : bin;

      float w_k = Wb[0], h_k = Hb[0], cw_k = cw[0], ch_k = ch[0];
      float d_k = d[0], d_kp1 = d[1];
#pragma unroll
      for (int i = 1; i < 16; ++i) {
        if (i == bin) {
          w_k = Wb[i]; h_k = Hb[i]; cw_k = cw[i]; ch_k = ch[i];
          d_k = d[i]; d_kp1 = d[i + 1];
        }
      }

      float xi = fminf(fmaxf((xc - cw_k) / w_k, 0.f), 1.f);
      float om = 1.f - xi;
      float num = h_k * (d_k * xi * xi + 2.f * xi * om);
      float den = d_k + (d_kp1 + d_k - 2.f) * xi * om;
      float yv = ch_k + num / (den + 1e-8f);
      float dnum = h_k * h_k * (d_kp1 * xi * xi + 2.f * xi * om + d_k * om * om);
      float ld = __logf(dnum / (den * den * w_k + 1e-8f) + 1e-8f);

      y_out[(size_t)(row0 + row) * DIM + 2 * t + 1] = yv;
      logacc += ld;
    }
    __syncthreads();   // rawT reused next iteration
  }

  // lane == row owns its full row's log-det: direct store, no reduction
  logdet_out[row0 + lane] = logacc;
}

// ---------------------------------------------------------------------------
extern "C" void kernel_launch(void* const* d_in, const int* in_sizes, int n_in,
                              void* d_out, int out_size, void* d_ws,
                              size_t ws_size, hipStream_t stream) {
  const float* x   = (const float*)d_in[0];
  const float* ctx = (const float*)d_in[1];
  const float* W1  = (const float*)d_in[2];
  const float* b1  = (const float*)d_in[3];
  const float* W2  = (const float*)d_in[4];
  const float* b2  = (const float*)d_in[5];
  const float* W3  = (const float*)d_in[6];
  const float* b3  = (const float*)d_in[7];

  float* y_out      = (float*)d_out;
  float* logdet_out = y_out + (size_t)BATCH * DIM;

  _Float16* W1t = (_Float16*)d_ws;                 //  512*128 halves
  _Float16* W2t = W1t + HID * INP;                 //  512*512
  _Float16* W3t = W2t + HID * HID;                 // 1536*512

  nscl_convert_weights<<<1024, 256, 0, stream>>>(W1, W2, W3, W1t, W2t, W3t);
  nscl_fused<<<NBLOCKS, WAVES * 32, DYN_LDS, stream>>>(
      x, ctx, b1, b2, b3, W1t, W2t, W3t, y_out, logdet_out);
}
